// MP_GI_50036368998507
// MI455X (gfx1250) — matrix operand order repaired and checked by reading, not compiled, not run on hardware
//
#include <hip/hip_runtime.h>
#include <hip/hip_bf16.h>

// ---------------- problem constants ----------------
#define NN   50000
#define EE   500000
#define DD   64
#define GG   50
#define NPG  1000        // NN / GG
#define KMSG 160         // 2D+1 = 129 padded to 5 chunks of 32
#define KUPD 128         // 2D    = 4 chunks of 32
#define PAIRNORM_EPS 1e-5f

typedef __attribute__((ext_vector_type(16))) __bf16 v16bf;
typedef __attribute__((ext_vector_type(8)))  float  v8f;

union ABf { uint4 u[2]; v16bf v; };

__device__ __forceinline__ unsigned short f2bf(float x) {
    unsigned int u = __float_as_uint(x);
    unsigned int r = (u + 0x7FFFu + ((u >> 16) & 1u)) >> 16;
    return (unsigned short)r;
}

__device__ __forceinline__ unsigned lds_addr_of(const void* p) {
    // generic address of LDS = {SHARED_BASE, lds_offset}; low 32 bits are the LDS address
    return (unsigned)(uintptr_t)p;
}

// async copy of one 128-byte row: global -> LDS (tracked by ASYNCcnt, retires in order)
#define ASYNC_ROW128(lds, ga)                                                                                   \
    do {                                                                                                        \
        asm volatile("global_load_async_to_lds_b128 %0, %1, off"             :: "v"(lds), "v"(ga) : "memory");  \
        asm volatile("global_load_async_to_lds_b128 %0, %1, off offset:16"   :: "v"(lds), "v"(ga) : "memory");  \
        asm volatile("global_load_async_to_lds_b128 %0, %1, off offset:32"   :: "v"(lds), "v"(ga) : "memory");  \
        asm volatile("global_load_async_to_lds_b128 %0, %1, off offset:48"   :: "v"(lds), "v"(ga) : "memory");  \
        asm volatile("global_load_async_to_lds_b128 %0, %1, off offset:64"   :: "v"(lds), "v"(ga) : "memory");  \
        asm volatile("global_load_async_to_lds_b128 %0, %1, off offset:80"   :: "v"(lds), "v"(ga) : "memory");  \
        asm volatile("global_load_async_to_lds_b128 %0, %1, off offset:96"   :: "v"(lds), "v"(ga) : "memory");  \
        asm volatile("global_load_async_to_lds_b128 %0, %1, off offset:112"  :: "v"(lds), "v"(ga) : "memory");  \
    } while (0)

#define WAIT_ASYNC_ALL()  asm volatile("s_wait_asynccnt 0x0"  ::: "memory")  // drain
#define WAIT_ASYNC_16()   asm volatile("s_wait_asynccnt 0x10" ::: "memory")  // keep prefetch batch in flight
#define WAIT_DS()         asm volatile("s_wait_dscnt 0x0"     ::: "memory")

// ---------------- small utility kernels ----------------
__global__ void zero_f32(float* __restrict__ p, int n) {
    int i = blockIdx.x * blockDim.x + threadIdx.x;
    if (i < n) p[i] = 0.0f;
}

// transpose+pad weights to bf16 [l][n][k]
__global__ void prep_weights(const float* __restrict__ msgW,  // [2,129,64]
                             const float* __restrict__ updW,  // [2,128,64]
                             unsigned short* __restrict__ msgWT, // [2,64,160]
                             unsigned short* __restrict__ updWT) // [2,64,128]
{
    int idx = blockIdx.x * blockDim.x + threadIdx.x;
    const int MSGT = 2 * 64 * KMSG;          // 20480
    const int UPDT = 2 * 64 * KUPD;          // 16384
    if (idx < MSGT) {
        int l = idx / (64 * KMSG);
        int r = idx % (64 * KMSG);
        int n = r / KMSG, k = r % KMSG;
        float v = (k < 2 * DD + 1) ? msgW[l * (2 * DD + 1) * 64 + k * 64 + n] : 0.0f;
        msgWT[idx] = f2bf(v);
    } else if (idx < MSGT + UPDT) {
        int j = idx - MSGT;
        int l = j / (64 * KUPD);
        int r = j % (64 * KUPD);
        int n = r / KUPD, k = r % KUPD;
        updWT[j] = f2bf(updW[l * KUPD * 64 + k * 64 + n]);
    }
}

// h0 = relu(h @ Wemb + bemb), write f32 + bf16
__global__ void embed_kernel(const float* __restrict__ hin,   // [N,5]
                             const float* __restrict__ Wemb,  // [5,64]
                             const float* __restrict__ bemb,  // [64]
                             float* __restrict__ hf,
                             unsigned short* __restrict__ hbf)
{
    int idx = blockIdx.x * blockDim.x + threadIdx.x;   // N*64
    if (idx >= NN * DD) return;
    int n = idx >> 6, c = idx & 63;
    float acc = bemb[c];
#pragma unroll
    for (int k = 0; k < 5; ++k) acc += hin[n * 5 + k] * Wemb[k * 64 + c];
    acc = fmaxf(acc, 0.0f);
    hf[idx]  = acc;
    hbf[idx] = f2bf(acc);
}

__global__ void dist_kernel(const float* __restrict__ pos,  // [N,2]
                            const int* __restrict__ eidx,   // [2,E]
                            float* __restrict__ dists)      // [E]
{
    int e = blockIdx.x * blockDim.x + threadIdx.x;
    if (e >= EE) return;
    int s = eidx[e], d = eidx[EE + e];
    float dx = pos[d * 2 + 0] - pos[s * 2 + 0];
    float dy = pos[d * 2 + 1] - pos[s * 2 + 1];
    dists[e] = sqrtf(dx * dx + dy * dy);
}

// ---------------- edge message GEMM + scatter ----------------
// one wave per 32-edge double-tile, double-buffered A staging:
// prefetch tile t+stride via global_load_async_to_lds_b128 into buf^1 while
// computing tile t from buf; s_wait_asynccnt 16 guarantees buf is ready
// (async loads retire in order) while the prefetch batch stays in flight.
__global__ void __launch_bounds__(128)
msg_kernel(const unsigned short* __restrict__ hbf,   // [N,64] bf16
           const unsigned short* __restrict__ Wt,    // [64,160] bf16 (layer slice)
           const float* __restrict__ bias,           // [64]  (layer slice)
           const float* __restrict__ dists,          // [E]
           const int* __restrict__ eidx,             // [2,E]
           float* __restrict__ aggr)                 // [N,64] (zeroed)
{
    __shared__ __align__(16) unsigned short sW[64 * KMSG];          // 20480 B
    __shared__ __align__(16) unsigned short sA[4][2][32 * KMSG];    // 81920 B
    const int tid  = threadIdx.x;
    const int lane = tid & 31;
    const int w    = tid >> 5;

    const unsigned sWb = lds_addr_of(&sW[0]);
    const unsigned sAb = lds_addr_of(&sA[w][0][0]);

    // stage W once per block (async, contiguous)
    for (int i = tid; i < (64 * KMSG) / 8; i += 128) {
        unsigned lds = sWb + (unsigned)i * 16u;
        unsigned long long ga = (unsigned long long)(uintptr_t)Wt + (unsigned long long)i * 16u;
        asm volatile("global_load_async_to_lds_b128 %0, %1, off" :: "v"(lds), "v"(ga) : "memory");
    }
    WAIT_ASYNC_ALL();
    __syncthreads();

    const int row   = lane & 15;
    const int khalf = lane >> 4;
    float bb[4];
#pragma unroll
    for (int nt = 0; nt < 4; ++nt) bb[nt] = bias[nt * 16 + row];

    const int nDT = EE / 32;                 // 15625 double-tiles
    const int gw = blockIdx.x * 4 + w;
    const int nw = gridDim.x * 4;

    // issues exactly 16 async loads per lane (2 row-copies x 8 x b128)
    auto stage = [&](int dtile, int b) {
        const int ebase = dtile * 32;
#pragma unroll
        for (int t = 0; t < 2; ++t) {
            int task = lane + t * 32;
            int e = task >> 1, sel = task & 1;
            int edge = ebase + e;
            int node = sel ? eidx[edge] : eidx[EE + edge];
            unsigned lds = sAb + (unsigned)(b * 32 * KMSG + e * KMSG + sel * DD) * 2u;
            unsigned long long ga = (unsigned long long)(uintptr_t)(hbf + (size_t)node * DD);
            ASYNC_ROW128(lds, ga);
            if (sel) {  // pad cols 128..159, then dist at col 128
                uint4 zz = {0u, 0u, 0u, 0u};
                uint4* z = (uint4*)(&sA[w][b][e * KMSG + 128]);
                z[0] = zz; z[1] = zz; z[2] = zz; z[3] = zz;
                sA[w][b][e * KMSG + 128] = f2bf(dists[edge]);
            }
        }
    };

    int buf = 0;
    if (gw < nDT) stage(gw, 0);

    for (int dt = gw; dt < nDT; dt += nw, buf ^= 1) {
        const int ebase = dt * 32;
        const int nxt = dt + nw;
        if (nxt < nDT) {
            stage(nxt, buf ^ 1);   // prefetch into other buffer
            WAIT_ASYNC_16();       // current buffer's 16 copies have retired
        } else {
            WAIT_ASYNC_ALL();      // drain
        }
        WAIT_DS();

        // ---- WMMAs: B fragment reused across both M-tiles
        v8f acc[2][4] = {{{}, {}, {}, {}}, {{}, {}, {}, {}}};
#pragma unroll
        for (int kc = 0; kc < 5; ++kc) {
            ABf a0, a1;
            const uint4* ap0 = (const uint4*)(&sA[w][buf][row * KMSG + kc * 32 + khalf * 8]);
            const uint4* ap1 = (const uint4*)(&sA[w][buf][(row + 16) * KMSG + kc * 32 + khalf * 8]);
            a0.u[0] = ap0[0]; a0.u[1] = ap0[2];
            a1.u[0] = ap1[0]; a1.u[1] = ap1[2];
#pragma unroll
            for (int nt = 0; nt < 4; ++nt) {
                ABf b;
                const uint4* bp = (const uint4*)(&sW[(nt * 16 + row) * KMSG + kc * 32 + khalf * 8]);
                b.u[0] = bp[0];
                b.u[1] = bp[2];
                acc[0][nt] = __builtin_amdgcn_wmma_f32_16x16x32_bf16(
                    false, a0.v, false, b.v, (short)0, acc[0][nt], false, false);
                acc[1][nt] = __builtin_amdgcn_wmma_f32_16x16x32_bf16(
                    false, a1.v, false, b.v, (short)0, acc[1][nt], false, false);
            }
        }

        // ---- bias + relu + atomic scatter (segment_sum by dst)
        const int rbase = khalf * 8;
#pragma unroll
        for (int mh = 0; mh < 2; ++mh) {
            int dsts[8];
#pragma unroll
            for (int r = 0; r < 8; ++r) dsts[r] = eidx[EE + ebase + mh * 16 + rbase + r];
#pragma unroll
            for (int nt = 0; nt < 4; ++nt) {
                int feat = nt * 16 + row;
#pragma unroll
                for (int r = 0; r < 8; ++r) {
                    float v = fmaxf(acc[mh][nt][r] + bb[nt], 0.0f);
                    atomicAdd(&aggr[(size_t)dsts[r] * DD + feat], v);
                }
            }
        }
    }
}

// ---------------- node update GEMM + pairnorm stats ----------------
__global__ void __launch_bounds__(128)
upd_kernel(const unsigned short* __restrict__ hbf,   // [N,64] bf16 (old h)
           const float* __restrict__ aggr,           // [N,64]
           const unsigned short* __restrict__ Wt,    // [64,128] bf16 (layer slice)
           const float* __restrict__ bias,           // [64]
           float* __restrict__ hf_out,               // [N,64] new h (pre-norm)
           float* __restrict__ colsum,               // [64]
           float* __restrict__ colsumsq)             // [64]
{
    __shared__ __align__(16) unsigned short sW[64 * KUPD];       // 16384 B
    __shared__ __align__(16) unsigned short sA[4][16 * KUPD];    // 16384 B
    const int tid  = threadIdx.x;
    const int lane = tid & 31;
    const int w    = tid >> 5;

    const unsigned sWb = lds_addr_of(&sW[0]);
    const unsigned sAb = lds_addr_of(&sA[w][0]);

    for (int i = tid; i < (64 * KUPD) / 8; i += 128) {
        unsigned lds = sWb + (unsigned)i * 16u;
        unsigned long long ga = (unsigned long long)(uintptr_t)Wt + (unsigned long long)i * 16u;
        asm volatile("global_load_async_to_lds_b128 %0, %1, off" :: "v"(lds), "v"(ga) : "memory");
    }
    WAIT_ASYNC_ALL();
    __syncthreads();

    const int row   = lane & 15;
    const int khalf = lane >> 4;
    float bb[4];
#pragma unroll
    for (int nt = 0; nt < 4; ++nt) bb[nt] = bias[nt * 16 + row];

    const int nTiles = NN / 16;
    const int gw = blockIdx.x * 4 + w;
    const int nw = gridDim.x * 4;

    for (int tile = gw; tile < nTiles; tile += nw) {
        const int nbase = tile * 16;
        { // stage A tile: cols 0..63 = h (async bf16 copy), 64..127 = bf16(aggr)
            int e = lane >> 1, sel = lane & 1;
            int node = nbase + e;
            if (sel == 0) {
                unsigned lds = sAb + (unsigned)(e * KUPD) * 2u;
                unsigned long long ga = (unsigned long long)(uintptr_t)(hbf + (size_t)node * DD);
                ASYNC_ROW128(lds, ga);
            } else {
                const float4* ar = (const float4*)(aggr + (size_t)node * DD);
                unsigned int* drow = (unsigned int*)(&sA[w][e * KUPD + DD]);
#pragma unroll
                for (int i = 0; i < 16; ++i) {
                    float4 f = ar[i];
                    drow[i * 2 + 0] = (unsigned)f2bf(f.x) | ((unsigned)f2bf(f.y) << 16);
                    drow[i * 2 + 1] = (unsigned)f2bf(f.z) | ((unsigned)f2bf(f.w) << 16);
                }
            }
        }
        WAIT_ASYNC_ALL();
        WAIT_DS();

        v8f acc[4] = {{}, {}, {}, {}};
#pragma unroll
        for (int kc = 0; kc < 4; ++kc) {
            ABf a;
            const uint4* ap = (const uint4*)(&sA[w][row * KUPD + kc * 32 + khalf * 8]);
            a.u[0] = ap[0];
            a.u[1] = ap[2];
#pragma unroll
            for (int nt = 0; nt < 4; ++nt) {
                ABf b;
                const uint4* bp = (const uint4*)(&sW[(nt * 16 + row) * KUPD + kc * 32 + khalf * 8]);
                b.u[0] = bp[0];
                b.u[1] = bp[2];
                acc[nt] = __builtin_amdgcn_wmma_f32_16x16x32_bf16(
                    false, a.v, false, b.v, (short)0, acc[nt], false, false);
            }
        }

        const int rbase = khalf * 8;
#pragma unroll
        for (int nt = 0; nt < 4; ++nt) {
            int feat = nt * 16 + row;
            float s = 0.0f, s2 = 0.0f;
#pragma unroll
            for (int r = 0; r < 8; ++r) {
                float v = fmaxf(acc[nt][r] + bb[nt], 0.0f);
                hf_out[(size_t)(nbase + rbase + r) * DD + feat] = v;
                s += v; s2 += v * v;
            }
            float os  = __shfl_xor(s, 16, 32);
            float os2 = __shfl_xor(s2, 16, 32);
            if (khalf == 0) {
                atomicAdd(&colsum[feat],   s + os);
                atomicAdd(&colsumsq[feat], s2 + os2);
            }
        }
    }
}

// stats layout: [0:64] colsum, [64:128] colsumsq, [128:192] mu, [192] inv_denom
__global__ void finalize_stats(float* __restrict__ stats) {
    __shared__ float red[64];
    int c = threadIdx.x;
    float mu = stats[c] * (1.0f / (float)NN);
    stats[128 + c] = mu;
    red[c] = stats[64 + c] - (float)NN * mu * mu;
    __syncthreads();
    for (int s = 32; s > 0; s >>= 1) {
        if (c < s) red[c] += red[c + s];
        __syncthreads();
    }
    if (c == 0) stats[192] = 1.0f / sqrtf(PAIRNORM_EPS + red[0] / (float)NN);
}

__global__ void normalize_kernel(float* __restrict__ hf,
                                 unsigned short* __restrict__ hbf,
                                 const float* __restrict__ stats)
{
    int idx = blockIdx.x * blockDim.x + threadIdx.x;
    if (idx >= NN * DD) return;
    int c = idx & 63;
    float v = (hf[idx] - stats[128 + c]) * stats[192];
    hf[idx]  = v;
    hbf[idx] = f2bf(v);
}

// segment_max over contiguous graphs (1000 nodes each)
__global__ void maxpool_kernel(const float* __restrict__ hf,
                               float* __restrict__ hg)   // [G,64]
{
    int g = blockIdx.x, c = threadIdx.x;
    float m = -3.402823466e38f;
    const float* p = hf + (size_t)g * NPG * DD + c;
    for (int i = 0; i < NPG; ++i) m = fmaxf(m, p[(size_t)i * DD]);
    hg[g * DD + c] = m;
}

// head: relu(hg@W1+b1) @ W2 + b2 -> out [G,2]
__global__ void __launch_bounds__(128)
final_mlp(const float* __restrict__ hg,
          const float* __restrict__ W1, const float* __restrict__ b1,
          const float* __restrict__ W2, const float* __restrict__ b2,
          float* __restrict__ out)
{
    __shared__ float shg[GG * DD];
    __shared__ float sh1[GG * DD];
    int tid = threadIdx.x;
    for (int i = tid; i < GG * DD; i += 128) shg[i] = hg[i];
    __syncthreads();
    if (tid < DD) {
        for (int g = 0; g < GG; ++g) {
            float acc = b1[tid];
            for (int k = 0; k < DD; ++k) acc += shg[g * DD + k] * W1[k * DD + tid];
            sh1[g * DD + tid] = fmaxf(acc, 0.0f);
        }
    }
    __syncthreads();
    if (tid < GG * 2) {
        int g = tid >> 1, o = tid & 1;
        float acc = b2[o];
        for (int k = 0; k < DD; ++k) acc += sh1[g * DD + k] * W2[k * 2 + o];
        out[g * 2 + o] = acc;
    }
}

// ---------------- host launcher ----------------
extern "C" void kernel_launch(void* const* d_in, const int* in_sizes, int n_in,
                              void* d_out, int out_size, void* d_ws, size_t ws_size,
                              hipStream_t stream) {
    const float* h    = (const float*)d_in[0];
    const float* pos  = (const float*)d_in[1];
    const int*   eidx = (const int*)d_in[2];
    // d_in[3] = batch (unused: graphs are contiguous blocks of 1000)
    const float* Wemb = (const float*)d_in[4];
    const float* bemb = (const float*)d_in[5];
    const float* msgW = (const float*)d_in[6];
    const float* msgb = (const float*)d_in[7];
    const float* updW = (const float*)d_in[8];
    const float* updb = (const float*)d_in[9];
    const float* W1   = (const float*)d_in[10];
    const float* b1   = (const float*)d_in[11];
    const float* W2   = (const float*)d_in[12];
    const float* b2   = (const float*)d_in[13];
    float* out = (float*)d_out;

    char* ws = (char*)d_ws;
    size_t off = 0;
    auto alloc = [&](size_t bytes) { char* p = ws + off; off = (off + bytes + 255) & ~(size_t)255; return p; };
    float*          hf     = (float*)alloc((size_t)NN * DD * 4);
    float*          aggr   = (float*)alloc((size_t)NN * DD * 4);
    unsigned short* hbf    = (unsigned short*)alloc((size_t)NN * DD * 2);
    float*          dists  = (float*)alloc((size_t)EE * 4);
    unsigned short* msgWT  = (unsigned short*)alloc((size_t)2 * 64 * KMSG * 2);
    unsigned short* updWT  = (unsigned short*)alloc((size_t)2 * 64 * KUPD * 2);
    float*          stats  = (float*)alloc(256 * 4);
    float*          hg     = (float*)alloc((size_t)GG * DD * 4);
    (void)ws_size;

    prep_weights<<<(2 * 64 * KMSG + 2 * 64 * KUPD + 255) / 256, 256, 0, stream>>>(msgW, updW, msgWT, updWT);
    embed_kernel<<<(NN * DD) / 256, 256, 0, stream>>>(h, Wemb, bemb, hf, hbf);
    dist_kernel<<<(EE + 255) / 256, 256, 0, stream>>>(pos, eidx, dists);

    for (int l = 0; l < 2; ++l) {
        zero_f32<<<(NN * DD) / 256, 256, 0, stream>>>(aggr, NN * DD);
        msg_kernel<<<512, 128, 0, stream>>>(hbf, msgWT + (size_t)l * 64 * KMSG,
                                            msgb + l * 64, dists, eidx, aggr);
        zero_f32<<<1, 128, 0, stream>>>(stats, 128);
        upd_kernel<<<256, 128, 0, stream>>>(hbf, aggr, updWT + (size_t)l * 64 * KUPD,
                                            updb + l * 64, hf, stats, stats + 64);
        finalize_stats<<<1, 64, 0, stream>>>(stats);
        normalize_kernel<<<(NN * DD) / 256, 256, 0, stream>>>(hf, hbf, stats);
    }

    maxpool_kernel<<<GG, DD, 0, stream>>>(hf, hg);
    final_mlp<<<1, 128, 0, stream>>>(hg, W1, b1, W2, b2, out);
}
